// ElmoAspectSent_10496900071999
// MI455X (gfx1250) — compile-verified, matrix operand order checked by reading, not run on hardware
//
#include <hip/hip_runtime.h>
#include <hip/hip_fp16.h>
#include <math.h>

typedef __attribute__((ext_vector_type(16))) _Float16 v16h;
typedef __attribute__((ext_vector_type(8)))  float    v8f;
typedef __attribute__((ext_vector_type(4)))  unsigned int u32x4;
typedef __attribute__((ext_vector_type(8)))  int      i32x8;
typedef __attribute__((ext_vector_type(4)))  int      i32x4;

#define B_   128
#define T_   128
#define E_   1024
#define MD_  50
#define IN_  1074
#define INP_ 1088      // IN padded to multiple of 32 (and of 64 for TDM slabs)
#define H_   256
#define G_   768       // 3H
#define G2_  1536      // both directions
#define D_   512
#define CO_  256
#define KC_  1536      // conv GEMM K = 3*512
#define NT_  16384     // B_*T_

#if __has_builtin(__builtin_amdgcn_tensor_load_to_lds) && \
    __has_builtin(__builtin_amdgcn_s_wait_tensorcnt)
#define HAVE_TDM 1
#else
#define HAVE_TDM 0
#endif

// ---------------- WMMA fragment helpers (wave32, 16x16x32 f16) ----------------
// A 16x32 f16 per-lane K pattern (ISA 7.12.2): K = e + (e>=8?8:0) + ((lane&16)?8:0)
__device__ __forceinline__ v16h frag_load(const _Float16* p, int koffs) {
  v16h f;
#pragma unroll
  for (int e = 0; e < 16; ++e) f[e] = p[e + ((e >= 8) ? 8 : 0) + koffs];
  return f;
}

__device__ __forceinline__ v16h frag_zero() {
  v16h f;
#pragma unroll
  for (int e = 0; e < 16; ++e) f[e] = (_Float16)0.f;
  return f;
}

__device__ __forceinline__ v8f wmma32(v16h a, v16h b, v8f c) {
  return __builtin_amdgcn_wmma_f32_16x16x32_f16(false, a, false, b, (short)0, c,
                                                false, false);
}

// ------------------------------- prep kernels --------------------------------
__global__ void prep_x_kernel(const float* __restrict__ sents,
                              const int* __restrict__ masks,
                              const float* __restrict__ memb,
                              _Float16* __restrict__ X) {
  const long n = (long)NT_ * INP_;
  for (long i = (long)blockIdx.x * blockDim.x + threadIdx.x; i < n;
       i += (long)gridDim.x * blockDim.x) {
    const int r = (int)(i / INP_), k = (int)(i % INP_);
    float v = 0.f;
    if (k < E_)        v = sents[(long)r * E_ + k];
    else if (k < IN_)  v = memb[masks[r] * MD_ + (k - E_)];
    X[i] = (_Float16)v;
  }
}

__global__ void prep_wih_kernel(const float* __restrict__ Wf,
                                const float* __restrict__ Wb,
                                _Float16* __restrict__ W) {
  const long n = (long)G2_ * INP_;
  for (long i = (long)blockIdx.x * blockDim.x + threadIdx.x; i < n;
       i += (long)gridDim.x * blockDim.x) {
    const int g = (int)(i / INP_), k = (int)(i % INP_);
    float v = 0.f;
    if (k < IN_) v = (g < G_) ? Wf[(long)g * IN_ + k] : Wb[(long)(g - G_) * IN_ + k];
    W[i] = (_Float16)v;
  }
}

__global__ void prep_whh_kernel(const float* __restrict__ Wf,
                                const float* __restrict__ Wb,
                                _Float16* __restrict__ W) {
  const long n = (long)2 * G_ * H_;
  for (long i = (long)blockIdx.x * blockDim.x + threadIdx.x; i < n;
       i += (long)gridDim.x * blockDim.x) {
    const long d = i / ((long)G_ * H_), rest = i % ((long)G_ * H_);
    W[i] = (_Float16)(d ? Wb[rest] : Wf[rest]);
  }
}

__global__ void prep_convw_kernel(const float* __restrict__ cw,
                                  _Float16* __restrict__ W) {
  const long n = (long)CO_ * KC_;
  for (long i = (long)blockIdx.x * blockDim.x + threadIdx.x; i < n;
       i += (long)gridDim.x * blockDim.x) {
    const int o = (int)(i / KC_), kk = (int)(i % KC_);
    const int dt = kk / D_, c = kk % D_;
    W[i] = (_Float16)cw[(long)o * (D_ * 3) + c * 3 + dt];   // conv_w[CO][D][3]
  }
}

// ---------------- GEMM: XG[T][B][1536] = X[16384][1088] * W^T + bih ----------
// Block = 32 rows x 256 cols. The 32x64 f16 A-slab per K-step is staged into
// LDS by the Tensor Data Mover (one TDM descriptor per step, issued by wave 0,
// tracked with TENSORcnt); all 8 waves consume it via DS loads.
__global__ void __launch_bounds__(256) gemm_xg_kernel(
    const _Float16* __restrict__ X, const _Float16* __restrict__ W,
    const float* __restrict__ bih_f, const float* __restrict__ bih_b,
    float* __restrict__ XG) {
  __shared__ _Float16 As[32 * 64];                 // 4 KB K-slab
  const int wave = threadIdx.x >> 5, lane = threadIdx.x & 31;
  const int mloc = wave >> 2, nloc = wave & 3;     // 2 Mtiles x 4 Ngroups per block
  const int mblk = blockIdx.x / 6, nblk = blockIdx.x % 6;
  const int nlo = lane & 15, koffs = (lane & 16) ? 8 : 0;
  const int row0 = mblk * 32;
  const int colbase = nblk * 256 + nloc * 64;
  v8f c[4] = {{}, {}, {}, {}};
  for (int k0 = 0; k0 < INP_; k0 += 64) {
    __syncthreads();                               // previous slab consumed
#if HAVE_TDM
    if (wave == 0) {
      const unsigned long long ga =
          (unsigned long long)(const void*)(X + (size_t)row0 * INP_ + k0);
      const unsigned lds = (unsigned)(unsigned long long)(void*)As;
      u32x4 g0;
      g0[0] = 1u;                                  // count=1, user mode
      g0[1] = lds;                                 // lds_addr
      g0[2] = (unsigned)ga;                        // global_addr[31:0]
      g0[3] = (unsigned)(ga >> 32) | (2u << 30);   // global_addr[56:32] | type=2
      i32x8 g1;
      g1[0] = 1 << 16;                             // data_size = 2 bytes
      g1[1] = (INP_ & 0xFFFF) << 16;               // tensor_dim0 (no clipping)
      g1[2] = (NT_ & 0xFFFF) << 16;                // tensor_dim1
      g1[3] = 64 << 16;                            // tile_dim0 = 64 (K)
      g1[4] = 32;                                  // tile_dim1 = 32 rows
      g1[5] = INP_;                                // tensor_dim0_stride = 1088
      g1[6] = 0;
      g1[7] = 0;
      const i32x4 z4 = {0, 0, 0, 0};
#if __clang_major__ >= 23
      const i32x8 z8 = {0, 0, 0, 0, 0, 0, 0, 0};
      __builtin_amdgcn_tensor_load_to_lds(g0, g1, z4, z4, z8, 0);
#else
      __builtin_amdgcn_tensor_load_to_lds(g0, g1, z4, z4, 0);
#endif
      __builtin_amdgcn_s_wait_tensorcnt(0);
    }
#else
    for (int i = threadIdx.x; i < 32 * 64; i += 256) {
      const int r = i >> 6, k = i & 63;
      As[i] = X[(size_t)(row0 + r) * INP_ + k0 + k];
    }
#endif
    __syncthreads();                               // slab published
#pragma unroll
    for (int kc = 0; kc < 2; ++kc) {
      const v16h a = frag_load(As + (size_t)(mloc * 16 + nlo) * 64 + kc * 32, koffs);
      const int kg = k0 + kc * 32;
#pragma unroll
      for (int nt = 0; nt < 4; ++nt) {
        const int ncol = colbase + nt * 16 + nlo;
        const v16h b = frag_load(W + (size_t)ncol * INP_ + kg, koffs);
        c[nt] = wmma32(a, b, c[nt]);
      }
    }
  }
#pragma unroll
  for (int nt = 0; nt < 4; ++nt) {
    const int col = colbase + nt * 16 + nlo;
    const float bias = (col < G_) ? bih_f[col] : bih_b[col - G_];
#pragma unroll
    for (int v = 0; v < 8; ++v) {
      const int r = row0 + mloc * 16 + v + koffs;  // r = b*T + t
      const int bb = r >> 7, tt = r & 127;
      XG[((size_t)tt * B_ + bb) * G2_ + col] = c[nt][v] + bias;
    }
  }
}

// ---------------- GRU: persistent single-workgroup recurrence ----------------
// 32 waves: wave w -> batch tile mb=w>>2; q=w&3 owns jt in [q*4, q*4+4),
// i.e. gate column tiles {jt, 16+jt, 32+jt} so the gate math is wave-local.
__global__ void __launch_bounds__(1024) gru_kernel(
    const float* __restrict__ XG, const _Float16* __restrict__ Whh,
    const float* __restrict__ bhh, const int* __restrict__ lens,
    float* __restrict__ h32, _Float16* __restrict__ h16,
    _Float16* __restrict__ ctx16, int dir) {
  const int tid = threadIdx.x, lane = tid & 31, wave = tid >> 5;
  for (int i = tid; i < B_ * H_; i += 1024) { h32[i] = 0.f; h16[i] = (_Float16)0.f; }
  __threadfence(); __syncthreads();
  const int mb = wave >> 2, q = wave & 3;
  const int nlo = lane & 15, koffs = (lane & 16) ? 8 : 0;
  for (int step = 0; step < T_; ++step) {
    const int t = dir ? (T_ - 1 - step) : step;
    const int cur = step & 1, nxt = cur ^ 1;
    const float*    hC   = h32 + (size_t)cur * B_ * H_;
    float*          hN   = h32 + (size_t)nxt * B_ * H_;
    const _Float16* hC16 = h16 + (size_t)cur * B_ * H_;
    _Float16*       hN16 = h16 + (size_t)nxt * B_ * H_;
    for (int jj = 0; jj < 4; ++jj) {
      const int jt = q * 4 + jj;
      v8f cr = {}, cz = {}, cn = {};
      for (int k0 = 0; k0 < H_; k0 += 32) {
        const v16h a  = frag_load(hC16 + (size_t)(mb * 16 + nlo) * H_ + k0, koffs);
        const v16h br = frag_load(Whh + (size_t)(jt * 16 + nlo) * H_ + k0, koffs);
        const v16h bz = frag_load(Whh + (size_t)(H_ + jt * 16 + nlo) * H_ + k0, koffs);
        const v16h bn = frag_load(Whh + (size_t)(2 * H_ + jt * 16 + nlo) * H_ + k0, koffs);
        cr = wmma32(a, br, cr);
        cz = wmma32(a, bz, cz);
        cn = wmma32(a, bn, cn);
      }
      const int j = jt * 16 + nlo;
      const float bhr = bhh[j], bhz = bhh[H_ + j], bhn = bhh[2 * H_ + j];
#pragma unroll
      for (int v = 0; v < 8; ++v) {
        const int b = mb * 16 + v + koffs;
        const float* gg = XG + ((size_t)t * B_ + b) * G2_ + dir * G_;
        const float r  = 1.f / (1.f + __expf(-(gg[j] + cr[v] + bhr)));
        const float z  = 1.f / (1.f + __expf(-(gg[H_ + j] + cz[v] + bhz)));
        const float nn = tanhf(gg[2 * H_ + j] + r * (cn[v] + bhn));
        const float hp = hC[(size_t)b * H_ + j];
        const float hv = (t < lens[b]) ? ((1.f - z) * nn + z * hp) : hp;
        hN[(size_t)b * H_ + j]   = hv;
        hN16[(size_t)b * H_ + j] = (_Float16)hv;
        ctx16[((size_t)b * T_ + t) * D_ + dir * H_ + j] = (_Float16)hv;
      }
    }
    __threadfence(); __syncthreads();
  }
}

// ---------------- Conv1d(k=3,pad=1)+ReLU as implicit WMMA GEMM ---------------
__global__ void __launch_bounds__(256) conv_gemm_kernel(
    const _Float16* __restrict__ ctx16, const _Float16* __restrict__ Wc,
    const float* __restrict__ convb, float* __restrict__ ctx2) {
  const int wave = threadIdx.x >> 5, lane = threadIdx.x & 31;
  const int job  = blockIdx.x * 8 + wave;           // 1024 Mtiles * 4 Ngroups
  const int mt = job >> 2, nw = job & 3;
  const int nlo = lane & 15, koffs = (lane & 16) ? 8 : 0;
  const int row = mt * 16 + nlo;                    // r = b*T + t (tile never crosses b)
  const int b = row >> 7, t = row & 127;
  v8f c[4] = {{}, {}, {}, {}};
  for (int k0 = 0; k0 < KC_; k0 += 32) {
    const int dt = k0 >> 9;                         // 512%32==0: chunk stays in one tap
    const int tt = t + dt - 1;
    v16h a;
    if (tt >= 0 && tt < T_) {
      const _Float16* ap = ctx16 + ((size_t)b * T_ + tt) * D_ + (k0 - dt * D_);
      __builtin_prefetch(ap + 32, 0, 1);            // global_prefetch for next chunk
      a = frag_load(ap, koffs);
    } else {
      a = frag_zero();
    }
#pragma unroll
    for (int nt = 0; nt < 4; ++nt) {
      const int ncol = nw * 64 + nt * 16 + nlo;
      const v16h bf = frag_load(Wc + (size_t)ncol * KC_ + k0, koffs);
      c[nt] = wmma32(a, bf, c[nt]);
    }
  }
#pragma unroll
  for (int nt = 0; nt < 4; ++nt) {
    const int col = nw * 64 + nt * 16 + nlo;
    const float bias = convb[col];
#pragma unroll
    for (int v = 0; v < 8; ++v) {
      const int r = mt * 16 + v + koffs;
      ctx2[(size_t)r * CO_ + col] = fmaxf(c[nt][v] + bias, 0.f);
    }
  }
}

// ------------------------------ small tail kernels ---------------------------
__global__ void feats_kernel(const float* __restrict__ ctx2,
                             const float* __restrict__ Wt,
                             const float* __restrict__ bt,
                             float* __restrict__ feats) {
  const int r = blockIdx.x * blockDim.x + threadIdx.x;
  if (r >= NT_) return;
  const float* c = ctx2 + (size_t)r * CO_;
  for (int k = 0; k < 4; ++k) {
    float acc = bt[k];
    for (int i = 0; i < CO_; ++i) acc += c[i] * Wt[k * CO_ + i];
    feats[r * 4 + k] = acc;
  }
}

__global__ void __launch_bounds__(128) crf_kernel(
    const float* __restrict__ feats, const int* __restrict__ lens,
    const float* __restrict__ trans, float* __restrict__ alpha,
    float* __restrict__ sp, float* __restrict__ spsum) {
  const int b = threadIdx.x;
  if (b >= B_) return;
  float tr[4][4];
  for (int i = 0; i < 4; ++i)
    for (int k = 0; k < 4; ++k) tr[i][k] = trans[i * 4 + k];
  const int len = lens[b];
  const float* fb = feats + (size_t)b * T_ * 4;
  float a[4];
  for (int k = 0; k < 4; ++k) { a[k] = fb[k]; alpha[((size_t)b * T_) * 4 + k] = a[k]; }
  for (int t = 1; t < T_; ++t) {
    if (t < len) {
      float an[4];
      for (int k = 0; k < 4; ++k) {
        const float v0 = a[0] + tr[0][k], v1 = a[1] + tr[1][k];
        const float v2 = a[2] + tr[2][k], v3 = a[3] + tr[3][k];
        const float m = fmaxf(fmaxf(v0, v1), fmaxf(v2, v3));
        an[k] = fb[t * 4 + k] + m +
                logf(__expf(v0 - m) + __expf(v1 - m) + __expf(v2 - m) + __expf(v3 - m));
      }
      for (int k = 0; k < 4; ++k) a[k] = an[k];
    }
    for (int k = 0; k < 4; ++k) alpha[((size_t)b * T_ + t) * 4 + k] = a[k];
  }
  float bt2[4] = {0.f, 0.f, 0.f, 0.f};
  float ssum = 0.f;
  for (int t = T_ - 1; t >= 0; --t) {
    float v[4];
    for (int k = 0; k < 4; ++k) v[k] = alpha[((size_t)b * T_ + t) * 4 + k] + bt2[k];
    const float m = fmaxf(fmaxf(v[0], v[1]), fmaxf(v[2], v[3]));
    const float den = __expf(v[0] - m) + __expf(v[1] - m) + __expf(v[2] - m) + __expf(v[3] - m);
    const float spv = (t < len) ? (__expf(v[1] - m) / den) : 0.f;
    sp[(size_t)b * T_ + t] = spv;
    ssum += spv;
    if (t >= 1 && t < len) {
      float bn[4];
      for (int i = 0; i < 4; ++i) {
        const float w0 = tr[i][0] + fb[t * 4 + 0] + bt2[0];
        const float w1 = tr[i][1] + fb[t * 4 + 1] + bt2[1];
        const float w2 = tr[i][2] + fb[t * 4 + 2] + bt2[2];
        const float w3 = tr[i][3] + fb[t * 4 + 3] + bt2[3];
        const float m2 = fmaxf(fmaxf(w0, w1), fmaxf(w2, w3));
        bn[i] = m2 + logf(__expf(w0 - m2) + __expf(w1 - m2) + __expf(w2 - m2) + __expf(w3 - m2));
      }
      for (int i = 0; i < 4; ++i) bt2[i] = bn[i];
    }
  }
  spsum[b] = ssum;
}

__global__ void __launch_bounds__(256) sentv_kernel(
    const float* __restrict__ sp, const float* __restrict__ ctx2,
    const float* __restrict__ spsum, float* __restrict__ sentv) {
  const int b = blockIdx.x, c = threadIdx.x;
  float acc = 0.f;
  for (int t = 0; t < T_; ++t)
    acc += sp[(size_t)b * T_ + t] * ctx2[((size_t)b * T_ + t) * CO_ + c];
  sentv[b * CO_ + c] = acc / (spsum[b] * 0.5f);
}

__global__ void __launch_bounds__(128) loss_kernel(
    const float* __restrict__ sentv, const float* __restrict__ W_lab,
    const float* __restrict__ b_lab, const int* __restrict__ labels,
    const float* __restrict__ spsum, const float* __restrict__ trans,
    float* __restrict__ out) {
  __shared__ float sh1[128], sh2[128];
  const int b = threadIdx.x;
  float s[3];
  for (int k = 0; k < 3; ++k) {
    float acc = b_lab[k];
    for (int c = 0; c < CO_; ++c) acc += sentv[b * CO_ + c] * W_lab[k * CO_ + c];
    s[k] = acc;
  }
  const float m = fmaxf(s[0], fmaxf(s[1], s[2]));
  const float lse = m + logf(__expf(s[0] - m) + __expf(s[1] - m) + __expf(s[2] - m));
  sh1[b] = s[labels[b]] - lse;
  sh2[b] = spsum[b];
  __syncthreads();
  if (b == 0) {
    float a1 = 0.f, a2 = 0.f;
    for (int i = 0; i < B_; ++i) { a1 += sh1[i]; a2 += sh2[i]; }
    const float pena = fmaxf(trans[4] - trans[0], 0.f) + fmaxf(trans[1] - trans[5], 0.f);
    out[0] = -a1 / (float)B_;
    out[1] = 0.1f * pena + 0.1f * (a2 / (float)B_);
  }
}

// --------------------------------- launcher ----------------------------------
extern "C" void kernel_launch(void* const* d_in, const int* in_sizes, int n_in,
                              void* d_out, int out_size, void* d_ws, size_t ws_size,
                              hipStream_t stream) {
  (void)in_sizes; (void)n_in; (void)out_size; (void)ws_size;
  const float* sents  = (const float*)d_in[0];
  const int*   masks  = (const int*)  d_in[1];
  const int*   labels = (const int*)  d_in[2];
  const int*   lens   = (const int*)  d_in[3];
  const float* memb   = (const float*)d_in[4];
  const float* Wih_f  = (const float*)d_in[5];
  const float* Whh_f  = (const float*)d_in[6];
  const float* bih_f  = (const float*)d_in[7];
  const float* bhh_f  = (const float*)d_in[8];
  const float* Wih_b  = (const float*)d_in[9];
  const float* Whh_b  = (const float*)d_in[10];
  const float* bih_b  = (const float*)d_in[11];
  const float* bhh_b  = (const float*)d_in[12];
  const float* conv_w = (const float*)d_in[13];
  const float* conv_b = (const float*)d_in[14];
  const float* W_tri  = (const float*)d_in[15];
  const float* b_tri  = (const float*)d_in[16];
  const float* trans  = (const float*)d_in[17];
  const float* W_lab  = (const float*)d_in[18];
  const float* b_lab  = (const float*)d_in[19];

  char* p = (char*)d_ws;
  auto take = [&](size_t bytes) -> char* {
    char* r = p;
    p += (bytes + 255) & ~(size_t)255;
    return r;
  };
  _Float16* X16   = (_Float16*)take((size_t)NT_ * INP_ * 2);
  _Float16* WIH16 = (_Float16*)take((size_t)G2_ * INP_ * 2);
  float*    XG    = (float*)   take((size_t)T_ * B_ * G2_ * 4);
  _Float16* WHH16 = (_Float16*)take((size_t)2 * G_ * H_ * 2);
  float*    H32   = (float*)   take((size_t)2 * B_ * H_ * 4);
  _Float16* H16   = (_Float16*)take((size_t)2 * B_ * H_ * 2);
  _Float16* CTX16 = (_Float16*)take((size_t)B_ * T_ * D_ * 2);
  _Float16* WC16  = (_Float16*)take((size_t)CO_ * KC_ * 2);
  float*    CTX2  = (float*)   take((size_t)B_ * T_ * CO_ * 4);
  float*    FEATS = (float*)   take((size_t)B_ * T_ * 4 * 4);
  float*    ALPHA = (float*)   take((size_t)B_ * T_ * 4 * 4);
  float*    SP    = (float*)   take((size_t)B_ * T_ * 4);
  float*    SPSUM = (float*)   take((size_t)B_ * 4);
  float*    SENTV = (float*)   take((size_t)B_ * CO_ * 4);

  prep_x_kernel<<<8192, 256, 0, stream>>>(sents, masks, memb, X16);
  prep_wih_kernel<<<4096, 256, 0, stream>>>(Wih_f, Wih_b, WIH16);
  prep_whh_kernel<<<1536, 256, 0, stream>>>(Whh_f, Whh_b, WHH16);
  prep_convw_kernel<<<1536, 256, 0, stream>>>(conv_w, WC16);

  gemm_xg_kernel<<<3072, 256, 0, stream>>>(X16, WIH16, bih_f, bih_b, XG);

  gru_kernel<<<1, 1024, 0, stream>>>(XG, WHH16,            bhh_f, lens, H32, H16, CTX16, 0);
  gru_kernel<<<1, 1024, 0, stream>>>(XG, WHH16 + G_ * H_,  bhh_b, lens, H32, H16, CTX16, 1);

  conv_gemm_kernel<<<512, 256, 0, stream>>>(CTX16, WC16, conv_b, CTX2);

  feats_kernel<<<64, 256, 0, stream>>>(CTX2, W_tri, b_tri, FEATS);
  crf_kernel<<<1, 128, 0, stream>>>(FEATS, lens, trans, ALPHA, SP, SPSUM);
  sentv_kernel<<<128, 256, 0, stream>>>(SP, CTX2, SPSUM, SENTV);
  loss_kernel<<<1, 128, 0, stream>>>(SENTV, W_lab, b_lab, labels, SPSUM, trans,
                                     (float*)d_out);
}
